// WDMPNNEncoder_81458349736430
// MI455X (gfx1250) — compile-verified, matrix operand order checked by reading, not run on hardware
//
#include <hip/hip_runtime.h>

#define N_ATOMS_   100000
#define N_BONDS_   400000
#define ATOM_FDIM_ 133
#define BOND_FDIM_ 147
#define HIDDEN_    300
#define HID_P_     304   // 19 * 16
#define KI_P_      160   // 147 -> 5 * 32
#define KH_P_      320   // 300 -> 10 * 32
#define KO_P_      448   // 433 -> 14 * 32
#define TILES_N_   19
#define LDSPAD_    8     // pad (elements) -> conflict-free ds_load_b128
#define NTHREADS_  320   // 10 waves

typedef __attribute__((ext_vector_type(16))) __bf16       v16bf;
typedef __attribute__((ext_vector_type(8)))  float        v8f;
typedef __attribute__((ext_vector_type(4)))  unsigned int u32x4;

union BFVec {
  v16bf v;
  unsigned short s[16];
  u32x4 q[2];
};

__device__ __forceinline__ unsigned short f2bf(float f) {
  union { float f; unsigned int u; } c; c.f = f;
  unsigned int u = c.u;
  u += 0x7FFFu + ((u >> 16) & 1u);   // round-to-nearest-even
  return (unsigned short)(u >> 16);
}

__device__ __forceinline__ float bf2f(unsigned short h) {
  union { unsigned int u; float f; } c;
  c.u = ((unsigned int)h) << 16;
  return c.f;
}

// Operand fragment layout (ISA 7.12.2, 16-bit): lanes 0-15 hold K {0..7,16..23},
// lanes 16-31 hold K {8..15,24..31}; two 16-byte loads per 32-K chunk.
__device__ __forceinline__ v16bf load_w(const unsigned short* __restrict__ W,
                                        int col, int Kp, int k0, int half) {
  BFVec r;
  const u32x4* p = (const u32x4*)(W + (size_t)col * Kp + k0 + half * 8);
  r.q[0] = p[0];
  r.q[1] = p[2];
  return r.v;
}

__device__ __forceinline__ v16bf load_a_lds(const unsigned short* As, int stride,
                                            int l, int k0, int half) {
  BFVec r;
  const u32x4* p = (const u32x4*)(As + l * stride + k0 + half * 8);
  r.q[0] = p[0];
  r.q[1] = p[2];
  return r.v;
}

// ---------------- weight fp32 -> padded bf16 ----------------
__global__ void wconv(const float* __restrict__ src, unsigned short* __restrict__ dst,
                      int K, int Kp, int total) {
  int idx = blockIdx.x * blockDim.x + threadIdx.x;
  if (idx >= total) return;
  int n = idx / Kp, k = idx - n * Kp;
  float v = (n < HIDDEN_ && k < K) ? src[n * K + k] : 0.0f;
  dst[idx] = f2bf(v);
}

// ---------------- zero fill (float4 granularity) ----------------
__global__ void zero_f4(float* __restrict__ p, int n4) {
  int i = blockIdx.x * blockDim.x + threadIdx.x;
  if (i < n4) ((float4*)p)[i] = make_float4(0.f, 0.f, 0.f, 0.f);
}

// ---- shared GEMM compute phase: wave w handles N tiles {w, w+10}; ----
// ---- epilogue fuses bias+relu+residual+store+segment-sum scatter.  ----
// Outputs: HoutF (fp32, stride outsF) and/or HoutB (bf16, stride HID_P_).
__device__ __forceinline__ void gemm_compute(const unsigned short* As, int stride, int Kp,
                                             const unsigned short* __restrict__ W,
                                             const float* __restrict__ bias,
                                             const unsigned short* __restrict__ HresB,
                                             float* __restrict__ HoutF, int outsF,
                                             unsigned short* __restrict__ HoutB,
                                             float* __restrict__ SMnext,
                                             const int* ldsDEST,
                                             int stripe) {
  int t = threadIdx.x;
  int w = t >> 5, lane = t & 31;
  int l = lane & 15, half = lane >> 4;
  int wuni = __builtin_amdgcn_readfirstlane(w);   // wave-uniform scalar
  int tn0 = wuni, tn1 = wuni + 10;
  v8f acc0 = {}, acc1 = {};
  if (wuni < 9) {                                  // two N tiles
    for (int kc = 0; kc < Kp / 32; ++kc) {
      int k0 = kc * 32;
      v16bf a  = load_a_lds(As, stride, l, k0, half);
      v16bf b0 = load_w(W, tn0 * 16 + l, Kp, k0, half);
      acc0 = __builtin_amdgcn_wmma_f32_16x16x32_bf16(false, a, false, b0,
                                                     (short)0, acc0, false, false);
      v16bf b1 = load_w(W, tn1 * 16 + l, Kp, k0, half);
      acc1 = __builtin_amdgcn_wmma_f32_16x16x32_bf16(false, a, false, b1,
                                                     (short)0, acc1, false, false);
    }
  } else {                                         // one N tile
    for (int kc = 0; kc < Kp / 32; ++kc) {
      int k0 = kc * 32;
      v16bf a  = load_a_lds(As, stride, l, k0, half);
      v16bf b0 = load_w(W, tn0 * 16 + l, Kp, k0, half);
      acc0 = __builtin_amdgcn_wmma_f32_16x16x32_bf16(false, a, false, b0,
                                                     (short)0, acc0, false, false);
    }
  }

  int dst[8];
  if (SMnext) {
#pragma unroll
    for (int r = 0; r < 8; ++r) dst[r] = ldsDEST[r + half * 8];
  }

  auto epilogue = [&](int tn, const v8f& acc) {
    int col = tn * 16 + l;
    if (col < HIDDEN_) {
      float bs = bias[col];
#pragma unroll
      for (int r = 0; r < 8; ++r) {
        int m = stripe * 16 + r + half * 8;
        float v = acc[r] + bs;
        if (HresB) v += bf2f(HresB[(size_t)m * HID_P_ + col]);
        v = v > 0.0f ? v : 0.0f;
        if (HoutF) HoutF[(size_t)m * outsF + col] = v;
        if (HoutB) HoutB[(size_t)m * HID_P_ + col] = f2bf(v);
        if (SMnext) atomicAdd(&SMnext[(size_t)dst[r] * HID_P_ + col], v);
      }
    }
  };
  epilogue(tn0, acc0);
  if (wuni < 9) epilogue(tn1, acc1);
}

// ---------------- H0 = relu(f_bonds @ Wi^T + bi), scatter fused ----------------
__global__ __launch_bounds__(NTHREADS_) void gemm_wi(const float* __restrict__ fb,
                                                     const int* __restrict__ b2a,
                                                     const int* __restrict__ b2revb,
                                                     const unsigned short* __restrict__ W,
                                                     const float* __restrict__ bias,
                                                     unsigned short* __restrict__ HoutB,
                                                     float* __restrict__ SMnext) {
  __shared__ __align__(16) unsigned short As[16 * (KI_P_ + LDSPAD_)];
  __shared__ int ldsDEST[16];
  const int stride = KI_P_ + LDSPAD_;
  int stripe = blockIdx.x;
  if (threadIdx.x < 16) {
    int row = stripe * 16 + threadIdx.x;
    ldsDEST[threadIdx.x] = b2a[b2revb[row]];
  }
  for (int e = threadIdx.x; e < 16 * KI_P_; e += NTHREADS_) {
    int r = e / KI_P_, k = e - r * KI_P_;
    int row = stripe * 16 + r;
    float v = (k < BOND_FDIM_) ? fb[(size_t)row * BOND_FDIM_ + k] : 0.0f;
    As[r * stride + k] = f2bf(v);
  }
  __syncthreads();
  gemm_compute(As, stride, KI_P_, W, bias, nullptr,
               nullptr, 0, HoutB, SMnext, ldsDEST, stripe);
}

// ---- Hnew = relu(Hin + (SM[b2a] - Hin[b2revb]) @ Wh^T + bh) ----
// Hin is bf16 (stride HID_P_); gather fused into fill, scatter fused into epilogue.
__global__ __launch_bounds__(NTHREADS_) void gemm_wh(const unsigned short* __restrict__ Hin,
                                                     const float* __restrict__ SM,
                                                     const int* __restrict__ b2a,
                                                     const int* __restrict__ b2revb,
                                                     const unsigned short* __restrict__ W,
                                                     const float* __restrict__ bias,
                                                     float* __restrict__ HoutF, int outsF,
                                                     unsigned short* __restrict__ HoutB,
                                                     float* __restrict__ SMnext) {
  __shared__ __align__(16) unsigned short As[16 * (KH_P_ + LDSPAD_)];
  __shared__ int ldsSA[16];
  __shared__ int ldsRB[16];
  __shared__ int ldsDEST[16];
  const int stride = KH_P_ + LDSPAD_;
  int stripe = blockIdx.x;
  if (threadIdx.x < 16) {
    int row = stripe * 16 + threadIdx.x;
    int rb = b2revb[row];
    ldsSA[threadIdx.x] = b2a[row];
    ldsRB[threadIdx.x] = rb;
    ldsDEST[threadIdx.x] = b2a[rb];
  }
  __syncthreads();
  for (int e = threadIdx.x; e < 16 * KH_P_; e += NTHREADS_) {
    int r = e / KH_P_, k = e - r * KH_P_;
    float v = 0.0f;
    if (k < HIDDEN_) {
      v = SM[(size_t)ldsSA[r] * HID_P_ + k]
          - bf2f(Hin[(size_t)ldsRB[r] * HID_P_ + k]);
    }
    As[r * stride + k] = f2bf(v);
  }
  __syncthreads();
  gemm_compute(As, stride, KH_P_, W, bias, Hin,
               HoutF, outsF, HoutB, SMnext, ldsDEST, stripe);
}

// ---- H_atoms = relu(concat(f_atoms, SM) @ Wo^T + bo), concat fused ----
__global__ __launch_bounds__(NTHREADS_) void gemm_wo(const float* __restrict__ fa,
                                                     const float* __restrict__ SM,
                                                     const unsigned short* __restrict__ W,
                                                     const float* __restrict__ bias,
                                                     float* __restrict__ Hout) {
  __shared__ __align__(16) unsigned short As[16 * (KO_P_ + LDSPAD_)];
  const int stride = KO_P_ + LDSPAD_;
  int stripe = blockIdx.x;
  for (int e = threadIdx.x; e < 16 * KO_P_; e += NTHREADS_) {
    int r = e / KO_P_, k = e - r * KO_P_;
    int row = stripe * 16 + r;
    float v;
    if (k < ATOM_FDIM_)                v = fa[(size_t)row * ATOM_FDIM_ + k];
    else if (k < ATOM_FDIM_ + HIDDEN_) v = SM[(size_t)row * HID_P_ + (k - ATOM_FDIM_)];
    else                               v = 0.0f;
    As[r * stride + k] = f2bf(v);
  }
  __syncthreads();
  gemm_compute(As, stride, KO_P_, W, bias, nullptr,
               Hout, HIDDEN_, nullptr, nullptr, nullptr, stripe);
}

extern "C" void kernel_launch(void* const* d_in, const int* in_sizes, int n_in,
                              void* d_out, int out_size, void* d_ws, size_t ws_size,
                              hipStream_t stream) {
  const float* f_atoms = (const float*)d_in[0];
  const float* f_bonds = (const float*)d_in[1];
  const int*   b2a     = (const int*)d_in[2];
  const int*   b2revb  = (const int*)d_in[3];
  const float* Wi_w = (const float*)d_in[4];
  const float* Wi_b = (const float*)d_in[5];
  const float* Wh_w = (const float*)d_in[6];
  const float* Wh_b = (const float*)d_in[7];
  const float* Wo_w = (const float*)d_in[8];
  const float* Wo_b = (const float*)d_in[9];

  float* out_atoms = (float*)d_out;
  float* out_bonds = out_atoms + (size_t)N_ATOMS_ * HIDDEN_;

  char* ws = (char*)d_ws;
  size_t off = 0;
  auto take = [&](size_t bytes) -> void* {
    void* p = ws + off;
    off += (bytes + 255) & ~(size_t)255;
    return p;
  };
  unsigned short* WiB = (unsigned short*)take((size_t)HID_P_ * KI_P_ * 2);
  unsigned short* WhB = (unsigned short*)take((size_t)HID_P_ * KH_P_ * 2);
  unsigned short* WoB = (unsigned short*)take((size_t)HID_P_ * KO_P_ * 2);
  unsigned short* H0  = (unsigned short*)take((size_t)N_BONDS_ * HID_P_ * 2);  // 243 MB
  unsigned short* H1  = (unsigned short*)take((size_t)N_BONDS_ * HID_P_ * 2);  // 243 MB
  float* SM0 = (float*)take((size_t)N_ATOMS_ * HID_P_ * 4);                    // 122 MB
  float* SM1 = (float*)take((size_t)N_ATOMS_ * HID_P_ * 4);                    // 122 MB

  dim3 b256(256);
  dim3 bG(NTHREADS_);
  wconv<<<(HID_P_ * KI_P_ + 255) / 256, b256, 0, stream>>>(Wi_w, WiB, BOND_FDIM_, KI_P_, HID_P_ * KI_P_);
  wconv<<<(HID_P_ * KH_P_ + 255) / 256, b256, 0, stream>>>(Wh_w, WhB, HIDDEN_, KH_P_, HID_P_ * KH_P_);
  wconv<<<(HID_P_ * KO_P_ + 255) / 256, b256, 0, stream>>>(Wo_w, WoB, ATOM_FDIM_ + HIDDEN_, KO_P_, HID_P_ * KO_P_);

  const int bond_stripes = N_BONDS_ / 16;   // 25000
  const int atom_stripes = N_ATOMS_ / 16;   // 6250
  const int sm4 = N_ATOMS_ * HID_P_ / 4;
  const int zblocks = (sm4 + 255) / 256;

  // H0 (bf16), scatter(H0) -> SM0
  zero_f4<<<zblocks, b256, 0, stream>>>(SM0, sm4);
  gemm_wi<<<bond_stripes, bG, 0, stream>>>(f_bonds, b2a, b2revb, WiB, Wi_b, H0, SM0);

  // depth iteration 1: (SM0, H0) -> H1 (bf16), scatter(H1) -> SM1
  zero_f4<<<zblocks, b256, 0, stream>>>(SM1, sm4);
  gemm_wh<<<bond_stripes, bG, 0, stream>>>(H0, SM0, b2a, b2revb, WhB, Wh_b,
                                           nullptr, 0, H1, SM1);

  // depth iteration 2: (SM1, H1) -> H2 (fp32, straight into d_out), scatter -> SM0
  zero_f4<<<zblocks, b256, 0, stream>>>(SM0, sm4);
  gemm_wh<<<bond_stripes, bG, 0, stream>>>(H1, SM1, b2a, b2revb, WhB, Wh_b,
                                           out_bonds, HIDDEN_, nullptr, SM0);

  // output layer: concat(f_atoms, SM0) @ Wo^T
  gemm_wo<<<atom_stripes, bG, 0, stream>>>(f_atoms, SM0, WoB, Wo_b, out_atoms);

  (void)in_sizes; (void)n_in; (void)out_size; (void)ws_size;
}
